// LLRNN_54039278519025
// MI455X (gfx1250) — matrix operand order repaired and checked by reading, not compiled, not run on hardware
//
#include <hip/hip_runtime.h>
#include <hip/hip_bf16.h>

// CDNA5 / gfx1250, wave32. Fused LLRNN: input projection + recurrence + dense
// head in one persistent-tile kernel using v_wmma_f32_16x16x32_f16.

typedef __attribute__((ext_vector_type(16))) _Float16     v16h;
typedef __attribute__((ext_vector_type(8)))  float        v8f;
typedef __attribute__((ext_vector_type(8)))  unsigned int v8u;

#define T_STEPS 256
#define IN_D    64
#define CELL    128
#define GATES   256
#define OUT_D   64
#define BM      16          // batch rows per workgroup (one 16-row M tile)
#define BATCH   4096

__device__ __forceinline__ v16h load_a_frag_lds(const _Float16* cb, int m, int k0) {
    // A fragment in this kernel's contiguous k order (same as B): K = k0..k0+7 (dw0-3) and k0+8..k0+15 (dw4-7)
    uint4 r0 = *(const uint4*)(cb + m * CELL + k0);
    uint4 r1 = *(const uint4*)(cb + m * CELL + k0 + 8);
    v8u av;
    av[0] = r0.x; av[1] = r0.y; av[2] = r0.z; av[3] = r0.w;
    av[4] = r1.x; av[5] = r1.y; av[6] = r1.z; av[7] = r1.w;
    return __builtin_bit_cast(v16h, av);
}

__global__ __launch_bounds__(256)
void llrnn_fused(const float* __restrict__ X,      // [BATCH][T][IN_D]
                 const float* __restrict__ Wk,     // [IN_D][GATES]
                 const float* __restrict__ Rk,     // [CELL][GATES]
                 const float* __restrict__ rbias,  // [GATES]
                 const float* __restrict__ Dw,     // [CELL][OUT_D]
                 const float* __restrict__ Db,     // [OUT_D]
                 float* __restrict__ Out)          // [BATCH][OUT_D]
{
    __shared__ _Float16 sC[2 * BM * CELL];   // double-buffered fp16 copy of c state
    __shared__ _Float16 sD[OUT_D * CELL];    // dense_w, transposed fp16 (filled at end)

    const int tid  = threadIdx.x;
    const int wave = tid >> 5;               // 0..7
    const int lane = tid & 31;
    const int ln   = lane & 15;              // column / row-in-tile index
    const int hi   = lane >> 4;              // half-wave select
    const int row0 = blockIdx.x * BM;

    // ---------------- one-time: weights -> register-resident fp16 B fragments
    // B 32x16 fragment: lane holds n = nt*16+ln, elements e=0..15 -> k = kc*32 + hi*16 + e
    v16h rB[2][4];                           // R fragments: [gate-half j][kchunk]
    v16h wB[2][2];                           // W fragments
    float bz[2];
    #pragma unroll
    for (int j = 0; j < 2; ++j) {
        const int nt = wave + 8 * j;         // j=0: sigmoid cols, j=1: tanh cols
        const int n  = nt * 16 + ln;
        bz[j] = rbias[n];
        #pragma unroll
        for (int kc = 0; kc < 4; ++kc) {
            v16h b;
            #pragma unroll
            for (int e = 0; e < 16; ++e)
                b[e] = (_Float16)Rk[(kc * 32 + hi * 16 + e) * GATES + n];
            rB[j][kc] = b;
        }
        #pragma unroll
        for (int kc = 0; kc < 2; ++kc) {
            v16h b;
            #pragma unroll
            for (int e = 0; e < 16; ++e)
                b[e] = (_Float16)Wk[(kc * 32 + hi * 16 + e) * GATES + n];
            wB[j][kc] = b;
        }
    }

    // zero initial state (registers + fp16 LDS copy, both buffers)
    v8f creg;                                // c for (m = hi*8+r, cell col wave*16+ln)
    #pragma unroll
    for (int r = 0; r < 8; ++r) creg[r] = 0.0f;
    for (int i = tid; i < 2 * BM * CELL; i += 256) sC[i] = (_Float16)0.0f;
    __syncthreads();

    // ---------------- sequential scan over T ----------------
    int p = 0;
    const float* xr = X + ((size_t)(row0 + ln) * T_STEPS) * IN_D;
    for (int t = 0; t < T_STEPS; ++t) {
        const _Float16* cb = sC + p * (BM * CELL);

        // A fragments: 4 chunks of c (fp16 LDS), 2 chunks of x_t (global f32->f16)
        v16h a_c[4];
        #pragma unroll
        for (int kc = 0; kc < 4; ++kc)
            a_c[kc] = load_a_frag_lds(cb, ln, kc * 32 + hi * 16);

        v16h a_x[2];
        const float* xt = xr + (size_t)t * IN_D;
        #pragma unroll
        for (int kc = 0; kc < 2; ++kc) {
            const int k0 = kc * 32 + hi * 16;
            float4 f0 = *(const float4*)(xt + k0);
            float4 f1 = *(const float4*)(xt + k0 + 4);
            float4 f2 = *(const float4*)(xt + k0 + 8);
            float4 f3 = *(const float4*)(xt + k0 + 12);
            v16h a;
            a[0]=(_Float16)f0.x; a[1]=(_Float16)f0.y; a[2]=(_Float16)f0.z; a[3]=(_Float16)f0.w;
            a[4]=(_Float16)f1.x; a[5]=(_Float16)f1.y; a[6]=(_Float16)f1.z; a[7]=(_Float16)f1.w;
            a[8]=(_Float16)f2.x; a[9]=(_Float16)f2.y; a[10]=(_Float16)f2.z; a[11]=(_Float16)f2.w;
            a[12]=(_Float16)f3.x; a[13]=(_Float16)f3.y; a[14]=(_Float16)f3.z; a[15]=(_Float16)f3.w;
            a_x[kc] = a;
        }
        // High-locality prefetch of next timestep's x row -> WGP-scope
        // global_prefetch_b8 (pull into all cache levels, shared by 8 waves).
        if (t + 1 < T_STEPS) __builtin_prefetch(xt + IN_D, 0, 3);

        // z_j = bias_j + c@R_j + x@W_j   (f32 accumulate, 12 WMMAs per wave per step)
        v8f z[2];
        #pragma unroll
        for (int j = 0; j < 2; ++j) {
            v8f acc;
            #pragma unroll
            for (int r = 0; r < 8; ++r) acc[r] = bz[j];
            #pragma unroll
            for (int kc = 0; kc < 4; ++kc)
                acc = __builtin_amdgcn_wmma_f32_16x16x32_f16(
                        false, a_c[kc], false, rB[j][kc], (short)0, acc, false, false);
            #pragma unroll
            for (int kc = 0; kc < 2; ++kc)
                acc = __builtin_amdgcn_wmma_f32_16x16x32_f16(
                        false, a_x[kc], false, wB[j][kc], (short)0, acc, false, false);
            z[j] = acc;
        }

        // gate update: f = sigmoid(z0), c = f*c + (1-f)*tanh(z1); all lane-local
        _Float16* cw = sC + (p ^ 1) * (BM * CELL);
        const int nc = wave * 16 + ln;       // cell column 0..127
        #pragma unroll
        for (int r = 0; r < 8; ++r) {
            float f  = 1.0f / (1.0f + __expf(-z[0][r]));
            float th = 1.0f - 2.0f / (__expf(2.0f * z[1][r]) + 1.0f);   // tanh
            float cn = f * creg[r] + (1.0f - f) * th;
            creg[r] = cn;
            cw[(hi * 8 + r) * CELL + nc] = (_Float16)cn;                // D-layout: m = hi*8+r
        }
        __syncthreads();
        p ^= 1;
    }

    // ---------------- dense head: out = c @ Dw + Db ----------------
    for (int i = tid; i < OUT_D * CELL; i += 256) {
        const int n = i / CELL, k = i % CELL;
        sD[i] = (_Float16)Dw[k * OUT_D + n];
    }
    __syncthreads();

    if (wave < 4) {                          // wave-uniform branch; EXEC all-1 inside
        const int ntO = wave;
        const _Float16* cb = sC + p * (BM * CELL);
        v8f acc;
        const float db = Db[ntO * 16 + ln];
        #pragma unroll
        for (int r = 0; r < 8; ++r) acc[r] = db;
        #pragma unroll
        for (int kc = 0; kc < 4; ++kc) {
            v16h a  = load_a_frag_lds(cb, ln, kc * 32 + hi * 16);
            v8u  bv = *(const v8u*)(sD + (ntO * 16 + ln) * CELL + kc * 32 + hi * 16);
            acc = __builtin_amdgcn_wmma_f32_16x16x32_f16(
                    false, a, false, __builtin_bit_cast(v16h, bv), (short)0, acc, false, false);
        }
        #pragma unroll
        for (int r = 0; r < 8; ++r)
            Out[(size_t)(row0 + hi * 8 + r) * OUT_D + ntO * 16 + ln] = acc[r];
    }
}

extern "C" void kernel_launch(void* const* d_in, const int* in_sizes, int n_in,
                              void* d_out, int out_size, void* d_ws, size_t ws_size,
                              hipStream_t stream) {
    (void)in_sizes; (void)n_in; (void)d_ws; (void)ws_size; (void)out_size;
    const float* X  = (const float*)d_in[0];
    const float* Wk = (const float*)d_in[1];
    const float* Rk = (const float*)d_in[2];
    const float* rb = (const float*)d_in[3];
    const float* Dw = (const float*)d_in[4];
    const float* Db = (const float*)d_in[5];
    float* Out = (float*)d_out;
    llrnn_fused<<<BATCH / BM, 256, 0, stream>>>(X, Wk, Rk, rb, Dw, Db, Out);
}
